// RNN_82257213653248
// MI455X (gfx1250) — matrix operand order repaired and checked by reading, not compiled, not run on hardware
//
#include <hip/hip_runtime.h>
#include <hip/hip_bf16.h>

#define Hdim 64
#define Bdim 256
#define Sdim 4096
#define BM   16      // batch rows per workgroup (one WMMA M tile)
#define NTHR 512     // 16 waves: one 16-col gate N-tile per wave

typedef __attribute__((ext_vector_type(16))) _Float16 v16h;
typedef __attribute__((ext_vector_type(8)))  float    v8f;

#if defined(__gfx1250__) && __has_builtin(__builtin_amdgcn_tanhf)
__device__ __forceinline__ float ftanh(float x) { return __builtin_amdgcn_tanhf(x); }
#else
__device__ __forceinline__ float ftanh(float x) {
    // tanh(x) = 1 - 2/(exp(2x)+1)
    return 1.0f - 2.0f * __builtin_amdgcn_rcpf(__expf(2.0f * x) + 1.0f);
}
#endif
// sigmoid(x) = 0.5*tanh(x/2) + 0.5  -> single TRANS op + fma on gfx1250
__device__ __forceinline__ float fsigmoid(float x) {
    return fmaf(ftanh(0.5f * x), 0.5f, 0.5f);
}

// B fragment (32x16, f16) for V_WMMA_F32_16X16X32_F16.
// K order used here (contiguous): VGPR v, lanes 0-15 hold K=2v,2v+1 (N=lane); lanes 16-31 hold K=16+2v,+1.
// => per lane: 16 contiguous K halves of weight row `ncol`, starting at kbase(+16 for hi half).
__device__ __forceinline__ v16h load_bfrag_f32(const float* __restrict__ W, int ncol, int kbase) {
    const float* p = W + ncol * Hdim + kbase;
    v16h b;
#pragma unroll
    for (int e = 0; e < 16; ++e) b[e] = (_Float16)p[e];
    return b;
}

// A fragment (16x32, f16) from an LDS row (row-major [16][72] f16, 16B aligned rows).
// Lanes 0-15: K = kb..kb+15 ; lanes 16-31 use kb += 16 (passed in).
__device__ __forceinline__ v16h load_afrag(const _Float16* __restrict__ row, int kb) {
    v16h a;
#pragma unroll
    for (int e = 0; e < 16; ++e) a[e] = row[kb + e];
    return a;
}

template <bool IS_L0>
__global__ __launch_bounds__(NTHR, 1)
void lstm_layer_kernel(const void*  __restrict__ in,     // L0: x[B,S,1] f32 ; else hs[B,S,H] f16 (in-place ok)
                       const float* __restrict__ Wih,    // [4H, din]
                       const float* __restrict__ Whh,    // [4H, H]
                       const float* __restrict__ bih,
                       const float* __restrict__ bhh,
                       _Float16*    __restrict__ hsOut,  // [B,S,H] f16 (nullptr => skip store)
                       float*       __restrict__ hnOut,  // [B,H]
                       float*       __restrict__ cnOut,  // [B,H]
                       const float* __restrict__ Wlin,   // [H] (only when yOut != nullptr)
                       const float* __restrict__ blin,   // [1]
                       float*       __restrict__ yOut)   // [B,S] f32 (nullptr => no fused head)
{
    __shared__ __align__(16) _Float16 hbuf[BM][72];   // h_t (A matrix for hh GEMM), padded
    __shared__ __align__(16) _Float16 xbuf[BM][72];   // prev-layer h_t (A for ih GEMM)
    __shared__ float gbuf[BM][260];                   // gate pre-activations [16][256], padded
    __shared__ float xcur[BM];                        // layer-0 scalar inputs
    __shared__ float wlin_s[Hdim + 1];                // fused-head weights (+bias)

    const int tid   = threadIdx.x;
    const int lane  = tid & 31;
    const int wv    = tid >> 5;          // 0..15 => gate N-tile
    const int bbase = blockIdx.x * BM;

    const int lhalf = lane >> 4;                 // 0/1 half of wave
    const int ncol  = wv * 16 + (lane & 15);     // gate column owned by this lane (N dim)
    const int arow  = lane & 15;                 // A-matrix row (M dim)
    const int kb    = lhalf * 16;                // A-fragment K base within 32-chunk

    // ---- persistent weight B-fragments in VGPRs (reused for all 4096 steps) ----
    v16h bhhf0 = load_bfrag_f32(Whh, ncol, 0  + 16 * lhalf);
    v16h bhhf1 = load_bfrag_f32(Whh, ncol, 32 + 16 * lhalf);
    v16h bihf0 = {}, bihf1 = {};
    float w0 = 0.0f;
    if (!IS_L0) {
        bihf0 = load_bfrag_f32(Wih, ncol, 0  + 16 * lhalf);
        bihf1 = load_bfrag_f32(Wih, ncol, 32 + 16 * lhalf);
    } else {
        w0 = Wih[ncol];                          // Wih is [256,1]
    }
    const float biasv = bih[ncol] + bhh[ncol];

    if (yOut != nullptr) {
        if (tid < Hdim) wlin_s[tid] = Wlin[tid];
        if (tid == Hdim) wlin_s[Hdim] = blin[0];
    }

    // h0 = 0
    for (int e = tid; e < BM * 72; e += NTHR) ((_Float16*)hbuf)[e] = (_Float16)0.0f;

    // persistent cell state: 2 cells per thread (tid, tid+512) over the 16x64 state
    float cst0 = 0.0f, cst1 = 0.0f;
    const int m0 = tid >> 6,          n0 = tid & 63;
    const int m1 = (tid + NTHR) >> 6, n1 = tid & 63;

    for (int t = 0; t < Sdim; ++t) {
        // ---- stage per-step inputs ----
        if (IS_L0) {
            if (tid < BM) {
                const float* x = (const float*)in;
                xcur[tid] = x[(size_t)(bbase + tid) * Sdim + t];
            }
        } else {
            if (tid < 128) {  // 16 rows x 8 segs, coalesced b128
                const int row = tid >> 3, seg = tid & 7;
                const _Float16* hp = (const _Float16*)in;
                const size_t off = ((size_t)(bbase + row) * Sdim + t) * Hdim + seg * 8;
                *(uint4*)(&xbuf[row][seg * 8]) = *(const uint4*)(hp + off);
                if (t + 1 < Sdim) __builtin_prefetch(hp + off + Hdim, 0, 0);
            }
        }
        __syncthreads();

        // ---- gates = bias (+ x*Wih) + h_prev @ Whh^T  via WMMA ----
        v8f acc;
#pragma unroll
        for (int r = 0; r < 8; ++r) {
            float v = biasv;
            if (IS_L0) v += xcur[r + 8 * lhalf] * w0;   // C layout: M = r + 8*lhalf
            acc[r] = v;
        }
        if (!IS_L0) {
            v16h a0 = load_afrag(&xbuf[arow][0], kb);
            acc = __builtin_amdgcn_wmma_f32_16x16x32_f16(false, a0, false, bihf0, (short)0, acc, false, false);
            v16h a1 = load_afrag(&xbuf[arow][0], 32 + kb);
            acc = __builtin_amdgcn_wmma_f32_16x16x32_f16(false, a1, false, bihf1, (short)0, acc, false, false);
        }
        {
            v16h h0 = load_afrag(&hbuf[arow][0], kb);
            acc = __builtin_amdgcn_wmma_f32_16x16x32_f16(false, h0, false, bhhf0, (short)0, acc, false, false);
            v16h h1 = load_afrag(&hbuf[arow][0], 32 + kb);
            acc = __builtin_amdgcn_wmma_f32_16x16x32_f16(false, h1, false, bhhf1, (short)0, acc, false, false);
        }

        // scatter D fragment to gate LDS: M = r + 8*lhalf, N = ncol
#pragma unroll
        for (int r = 0; r < 8; ++r) gbuf[r + 8 * lhalf][ncol] = acc[r];
        __syncthreads();

        // ---- elementwise LSTM cell (gate order i,f,g,o), 2 cells/thread ----
        {
            float gi = gbuf[m0][n0], gf = gbuf[m0][64 + n0];
            float gg = gbuf[m0][128 + n0], go = gbuf[m0][192 + n0];
            cst0 = fsigmoid(gf) * cst0 + fsigmoid(gi) * ftanh(gg);
            float h = fsigmoid(go) * ftanh(cst0);
            hbuf[m0][n0] = (_Float16)h;
            if (t == Sdim - 1) {
                hnOut[(size_t)(bbase + m0) * Hdim + n0] = h;
                cnOut[(size_t)(bbase + m0) * Hdim + n0] = cst0;
            }
        }
        {
            float gi = gbuf[m1][n1], gf = gbuf[m1][64 + n1];
            float gg = gbuf[m1][128 + n1], go = gbuf[m1][192 + n1];
            cst1 = fsigmoid(gf) * cst1 + fsigmoid(gi) * ftanh(gg);
            float h = fsigmoid(go) * ftanh(cst1);
            hbuf[m1][n1] = (_Float16)h;
            if (t == Sdim - 1) {
                hnOut[(size_t)(bbase + m1) * Hdim + n1] = h;
                cnOut[(size_t)(bbase + m1) * Hdim + n1] = cst1;
            }
        }
        __syncthreads();

        // ---- coalesced f16 store of h_t for the next layer ----
        if (hsOut != nullptr) {
            if (tid < 128) {
                const int row = tid >> 3, seg = tid & 7;
                *(uint4*)(hsOut + ((size_t)(bbase + row) * Sdim + t) * Hdim + seg * 8) =
                    *(const uint4*)(&hbuf[row][seg * 8]);
            }
        }
        // ---- fused linear head (last layer): y[b,t] = h_t . Wlin + blin ----
        if (yOut != nullptr) {
            if (tid < BM) {
                float a = wlin_s[Hdim];
#pragma unroll
                for (int k = 0; k < Hdim; ++k) a += (float)hbuf[tid][k] * wlin_s[k];
                yOut[(size_t)(bbase + tid) * Sdim + t] = a;
            }
        }
    }
}

extern "C" void kernel_launch(void* const* d_in, const int* in_sizes, int n_in,
                              void* d_out, int out_size, void* d_ws, size_t ws_size,
                              hipStream_t stream) {
    const float* x    = (const float*)d_in[0];
    const float* Wih0 = (const float*)d_in[1];
    const float* Whh0 = (const float*)d_in[2];
    const float* bih0 = (const float*)d_in[3];
    const float* bhh0 = (const float*)d_in[4];
    const float* Wih1 = (const float*)d_in[5];
    const float* Whh1 = (const float*)d_in[6];
    const float* bih1 = (const float*)d_in[7];
    const float* bhh1 = (const float*)d_in[8];
    const float* Wih2 = (const float*)d_in[9];
    const float* Whh2 = (const float*)d_in[10];
    const float* bih2 = (const float*)d_in[11];
    const float* bhh2 = (const float*)d_in[12];
    const float* Wlin = (const float*)d_in[13];
    const float* blin = (const float*)d_in[14];

    float* y  = (float*)d_out;                              // [B,S,1]
    float* hn = (float*)d_out + (size_t)Bdim * Sdim;        // [3,B,H]
    float* cn = hn + (size_t)3 * Bdim * Hdim;               // [3,B,H]

    _Float16* hs = (_Float16*)d_ws;  // [B,S,H] f16, 128 MiB, updated in-place across layers

    const size_t bh = (size_t)Bdim * Hdim;
    dim3 grid(Bdim / BM), block(NTHR);
    lstm_layer_kernel<true ><<<grid, block, 0, stream>>>((const void*)x,  Wih0, Whh0, bih0, bhh0,
                                                         hs, hn + 0 * bh, cn + 0 * bh,
                                                         nullptr, nullptr, nullptr);
    lstm_layer_kernel<false><<<grid, block, 0, stream>>>((const void*)hs, Wih1, Whh1, bih1, bhh1,
                                                         hs, hn + 1 * bh, cn + 1 * bh,
                                                         nullptr, nullptr, nullptr);
    lstm_layer_kernel<false><<<grid, block, 0, stream>>>((const void*)hs, Wih2, Whh2, bih2, bhh2,
                                                         nullptr, hn + 2 * bh, cn + 2 * bh,
                                                         Wlin, blin, y);
}
